// HVAE_55490977464568
// MI455X (gfx1250) — matrix operand order repaired and checked by reading, not compiled, not run on hardware
//
#include <hip/hip_runtime.h>
#include <hip/hip_bf16.h>
#include <stdint.h>

#define B_   8192
#define BSH  13          // log2(B_)
#define BMK  8191        // B_ - 1
#define SYM  32
#define H_   128
#define Z_   128
#define NN   31
#define G3   (3 * H_)   // 384, stacked [r;z;n] gates

typedef __attribute__((ext_vector_type(16))) _Float16 v16h;
typedef __attribute__((ext_vector_type(8)))  _Float16 v8h;
typedef __attribute__((ext_vector_type(8)))  float    v8f;

__device__ __forceinline__ float sigf_(float x) { return 1.0f / (1.0f + __expf(-x)); }
// row map: batched row r -> physical row, node-slab granularity is always B_ rows.
// stride==B_ -> identity; stride==2*B_ -> stride-2 node slabs (children of contiguous parents).
__device__ __forceinline__ int rmap_(int r, int stride) {
    return (r >> BSH) * stride + (r & BMK);
}

// ---------------------------------------------------------------------------
// WMMA GEMM: C[M,N] (f32) = A[M,K](f16, row-mapped) @ W[N,K]^T (f16) [+ bias][+= C]
// block = 256 thr = 8 waves; block tile 128(M) x 16*NT(N); wave tile 16 x 16*NT.
// W N-tile staged in LDS once (row padded +8 halves -> conflict-free b128 reads).
// K compile-time (32 or 128) -> K-loop fully unrolled, A loads hoisted.
// ACC/BIAS compile-time -> branch-free epilogue; N is an exact multiple of 16*NT
// for every dispatch, so no edge guards are needed anywhere.
// M % 128 == 0 guaranteed by caller; grid sized exactly.
// ---------------------------------------------------------------------------
template <int K, int NT, bool ACC, bool BIAS>
__global__ __launch_bounds__(256)
void k_gemm16(const _Float16* __restrict__ A, const _Float16* __restrict__ W,
              const float* __restrict__ bias, float* __restrict__ C,
              int M, int N, int aStr, int cStr)
{
    constexpr int LDK = K + 8;              // padded LDS row stride (halves)
    __shared__ _Float16 sW[64 * 136];       // max: NT=4, K=128 -> 64*136*2 = 17408 B

    const int tid   = threadIdx.x;
    const int lane  = tid & 31;
    const int wave  = tid >> 5;
    const int mBase = blockIdx.y * 128 + wave * 16;
    const int nBase = blockIdx.x * (16 * NT);
    const int ml = lane & 15;
    const int hi = lane >> 4;

    // ---- stage W tile [nBase, nBase+16*NT) x K into LDS ----
    constexpr int CH = K / 8;               // b128 chunks per row
    for (int idx = tid; idx < 16 * NT * CH; idx += 256) {
        int row = idx / CH, kc = (idx % CH) * 8;
        *(v8h*)(sW + row * LDK + kc) =
            *(const v8h*)(W + (size_t)(nBase + row) * K + kc);
    }
    __syncthreads();

    const int aBase = rmap_(mBase, aStr);   // shift+and, once per wave
    const int cBase = rmap_(mBase, cStr);
    const _Float16* aptr = A + (size_t)(aBase + ml) * K + 8 * hi;

    v8f acc[NT];
#pragma unroll
    for (int j = 0; j < NT; ++j) acc[j] = v8f{};

#pragma unroll
    for (int kb = 0; kb < K; kb += 32) {
        // A 16x32 fragment: halves 0..7 -> K = kb+8*hi.. ; 8..15 -> K = kb+16+8*hi..
        v8h a0 = *(const v8h*)(aptr + kb);
        v8h a1 = *(const v8h*)(aptr + kb + 16);
        v16h af;
#pragma unroll
        for (int i = 0; i < 8; ++i) { af[i] = a0[i]; af[i + 8] = a1[i]; }
#pragma unroll
        for (int j = 0; j < NT; ++j) {
            // B 32x16 fragment from LDS: lane col = ml, halves 0..7 -> K = kb+8*hi.. ; 8..15 -> K = kb+16+8*hi..
            const _Float16* bp = sW + (j * 16 + ml) * LDK + kb + 8 * hi;
            v8h b0 = *(const v8h*)(bp);
            v8h b1 = *(const v8h*)(bp + 16);
            v16h bf;
#pragma unroll
            for (int i = 0; i < 8; ++i) { bf[i] = b0[i]; bf[i + 8] = b1[i]; }
            acc[j] = __builtin_amdgcn_wmma_f32_16x16x32_f16(
                false, af, false, bf, (short)0, acc[j], false, false);
        }
    }

    // D layout: lane col = ml, row = v + 8*hi
    // one base-address multiply per tile, then pointer-stepped rows
#pragma unroll
    for (int j = 0; j < NT; ++j) {
        int n = nBase + j * 16 + ml;
        float bv = BIAS ? bias[n] : 0.0f;
        float* cp = C + (size_t)(cBase + 8 * hi) * N + n;
#pragma unroll
        for (int v = 0; v < 8; ++v) {
            float val = acc[j][v] + bv;
            if (ACC) val += *cp;
            *cp = val;
            cp += N;
        }
    }
}

// ---------------------------------------------------------------------------
// elementwise kernels
// ---------------------------------------------------------------------------
__global__ void k_f2h(const float* __restrict__ s, _Float16* __restrict__ d, long n)
{
    long i = (long)blockIdx.x * 256 + threadIdx.x;
    if (i < n) d[i] = (_Float16)s[i];
}

__global__ void k_copy16(const _Float16* __restrict__ s, _Float16* __restrict__ d,
                         int M, int Cc, int sStr, int dStr)
{
    long i = (long)blockIdx.x * 256 + threadIdx.x;
    if (i >= (long)M * Cc) return;
    int r = (int)(i / Cc), c = (int)(i % Cc);
    d[(size_t)rmap_(r, dStr) * Cc + c] = s[(size_t)rmap_(r, sStr) * Cc + c];
}

// leaf encoder node: h = (1-sig(iz+bhz)) * tanh(inn+bhn) * mask   (h_sum = s = 0)
__global__ void k_enc_leaf(const float* __restrict__ Xi, const float* __restrict__ ebh,
                           const float* __restrict__ masks, _Float16* __restrict__ h,
                           int M, int nodeBase)
{
    long i = (long)blockIdx.x * 256 + threadIdx.x;
    if (i >= (long)M * H_) return;
    int r = (int)(i / H_), c = (int)(i % H_);
    const float* xr = Xi + (size_t)r * G3;
    float zg = sigf_(xr[H_ + c] + ebh[H_ + c]);
    float nn = tanhf(xr[2 * H_ + c] + ebh[2 * H_ + c]);
    float mk = masks[(size_t)(nodeBase + (r >> BSH)) * B_ + (r & BMK)];
    h[(size_t)r * H_ + c] = (_Float16)((1.0f - zg) * nn * mk);
}

// s = sig(ir+RL)*hL + sig(ir+RR)*hR  -> f16
__global__ void k_enc_s(const float* __restrict__ Xi, const float* __restrict__ RL,
                        const float* __restrict__ RR, const _Float16* __restrict__ hL,
                        const _Float16* __restrict__ hR, _Float16* __restrict__ s16,
                        int M, int hStr)
{
    long i = (long)blockIdx.x * 256 + threadIdx.x;
    if (i >= (long)M * H_) return;
    int r = (int)(i / H_), c = (int)(i % H_);
    float ir = Xi[(size_t)r * G3 + c];
    int g = rmap_(r, hStr);
    float hl = (float)hL[(size_t)g * H_ + c];
    float hr = (float)hR[(size_t)g * H_ + c];
    float a = sigf_(ir + RL[(size_t)r * H_ + c]);
    float b = sigf_(ir + RR[(size_t)r * H_ + c]);
    s16[(size_t)r * H_ + c] = (_Float16)(a * hl + b * hr);
}

// h = ((1-z)*n + z*(hL+hR)) * mask ;  z=sig(iz+BZ), n=tanh(inn+BN)
__global__ void k_enc_h(const float* __restrict__ Xi, const float* __restrict__ BZ,
                        const float* __restrict__ BN, const _Float16* __restrict__ hL,
                        const _Float16* __restrict__ hR, const float* __restrict__ masks,
                        _Float16* __restrict__ hOut, int M, int nodeBase, int hStr)
{
    long i = (long)blockIdx.x * 256 + threadIdx.x;
    if (i >= (long)M * H_) return;
    int r = (int)(i / H_), c = (int)(i % H_);
    float z = sigf_(Xi[(size_t)r * G3 + H_ + c] + BZ[(size_t)r * H_ + c]);
    float n = tanhf(Xi[(size_t)r * G3 + 2 * H_ + c] + BN[(size_t)r * H_ + c]);
    int g = rmap_(r, hStr);
    float hsum = (float)hL[(size_t)g * H_ + c] + (float)hR[(size_t)g * H_ + c];
    float mk = masks[(size_t)(nodeBase + (r >> BSH)) * B_ + (r & BMK)];
    hOut[(size_t)r * H_ + c] = (_Float16)(((1.0f - z) * n + z * hsum) * mk);
}

__global__ void k_reparam(const float* __restrict__ mu, const float* __restrict__ lv,
                          const float* __restrict__ eps, _Float16* __restrict__ z16, long n)
{
    long i = (long)blockIdx.x * 256 + threadIdx.x;
    if (i < n) z16[i] = (_Float16)(mu[i] + eps[i] * __expf(0.5f * lv[i]));
}

// softmax over rows of 32 (one thread per row), row-mapped in/out
__global__ void k_softmax32(const float* __restrict__ pred, _Float16* __restrict__ probs,
                            int M, int str)
{
    int r = blockIdx.x * 256 + threadIdx.x;
    if (r >= M) return;
    int g = rmap_(r, str);
    const float* p = pred + (size_t)g * SYM;
    _Float16* o = probs + (size_t)g * SYM;
    float mx = p[0];
#pragma unroll
    for (int i = 1; i < SYM; ++i) mx = fmaxf(mx, p[i]);
    float s = 0.0f;
#pragma unroll
    for (int i = 0; i < SYM; ++i) s += __expf(p[i] - mx);
    float inv = 1.0f / s;
#pragma unroll
    for (int i = 0; i < SYM; ++i) o[i] = (_Float16)(__expf(p[i] - mx) * inv);
}

// GRU combine: r=sig(ir+hr) z=sig(iz+hz) n=tanh(inn+r*hn) ; out=(1-z)*n+z*ha
__global__ void k_gru(const float* __restrict__ I, const float* __restrict__ Hh,
                      const _Float16* __restrict__ ha, _Float16* __restrict__ outp,
                      int M, int haStr, int oStr)
{
    long i = (long)blockIdx.x * 256 + threadIdx.x;
    if (i >= (long)M * H_) return;
    int r = (int)(i / H_), c = (int)(i % H_);
    size_t b = (size_t)r * G3 + c;
    float rr = sigf_(I[b] + Hh[b]);
    float zz = sigf_(I[b + H_] + Hh[b + H_]);
    float nn = tanhf(I[b + 2 * H_] + rr * Hh[b + 2 * H_]);
    float hv = (float)ha[(size_t)rmap_(r, haStr) * H_ + c];
    outp[(size_t)rmap_(r, oStr) * H_ + c] = (_Float16)((1.0f - zz) * nn + zz * hv);
}

// fraternal GRU with h==0: hn terms are just fra_bh ; hf=(1-z)*n
__global__ void k_fra(const float* __restrict__ I, const float* __restrict__ fbh,
                      _Float16* __restrict__ hf, int M)
{
    long i = (long)blockIdx.x * 256 + threadIdx.x;
    if (i >= (long)M * H_) return;
    int r = (int)(i / H_), c = (int)(i % H_);
    size_t b = (size_t)r * G3 + c;
    float rr = sigf_(I[b] + fbh[c]);
    float zz = sigf_(I[b + H_] + fbh[H_ + c]);
    float nn = tanhf(I[b + 2 * H_] + rr * fbh[2 * H_ + c]);
    hf[(size_t)r * H_ + c] = (_Float16)((1.0f - zz) * nn);
}

__global__ void k_tanh16(const float* __restrict__ U, _Float16* __restrict__ outp,
                         int M, int oStr)
{
    long i = (long)blockIdx.x * 256 + threadIdx.x;
    if (i >= (long)M * H_) return;
    int r = (int)(i / H_), c = (int)(i % H_);
    outp[(size_t)rmap_(r, oStr) * H_ + c] = (_Float16)tanhf(U[(size_t)r * H_ + c]);
}

// ---------------------------------------------------------------------------
extern "C" void kernel_launch(void* const* d_in, const int* in_sizes, int n_in,
                              void* d_out_, int out_size, void* d_ws, size_t ws_size,
                              hipStream_t stream)
{
    (void)in_sizes; (void)n_in; (void)out_size; (void)ws_size;

    const float* targets = (const float*)d_in[0];
    const float* masks   = (const float*)d_in[1];
    const float* eps     = (const float*)d_in[2];
    const float* enc_Wi  = (const float*)d_in[3];
    const float* enc_bi  = (const float*)d_in[4];
    const float* enc_Wh  = (const float*)d_in[5];
    const float* enc_bh  = (const float*)d_in[6];
    const float* mu_W    = (const float*)d_in[7];
    const float* mu_b    = (const float*)d_in[8];
    const float* lv_W    = (const float*)d_in[9];
    const float* lv_b    = (const float*)d_in[10];
    const float* z2h_W   = (const float*)d_in[11];
    const float* z2h_b   = (const float*)d_in[12];
    const float* h2o_W   = (const float*)d_in[13];
    const float* h2o_b   = (const float*)d_in[14];
    const float* anc_Wi  = (const float*)d_in[15];
    const float* anc_bi  = (const float*)d_in[16];
    const float* anc_Wh  = (const float*)d_in[17];
    const float* anc_bh  = (const float*)d_in[18];
    const float* fra_Wi  = (const float*)d_in[19];
    const float* fra_bi  = (const float*)d_in[20];
    /* d_in[21] fra_Wh unused: fraternal GRU always runs on h==0 */
    const float* fra_bh  = (const float*)d_in[22];
    const float* ua_W    = (const float*)d_in[23];
    const float* ua_b    = (const float*)d_in[24];
    const float* uf_W    = (const float*)d_in[25];
    const float* uf_b    = (const float*)d_in[26];

    float* outMu   = (float*)d_out_;
    float* outLv   = outMu + (size_t)B_ * Z_;
    float* outPred = outMu + 2 * (size_t)B_ * Z_;   // [NN, B, SYM]

    uint8_t* ws = (uint8_t*)d_ws;
    size_t off = 0;
    auto take = [&](size_t bytes) -> uint8_t* {
        off = (off + 255) & ~(size_t)255;
        uint8_t* p = ws + off; off += bytes; return p;
    };
    auto takeH = [&](size_t e) -> _Float16* { return (_Float16*)take(e * 2); };
    auto takeF = [&](size_t e) -> float*    { return (float*)take(e * 4); };

    // f16 weight copies (W stored [N,K] row-major == torch convention: y = x @ W^T)
    _Float16* w_encWi = takeH((size_t)G3 * SYM);
    _Float16* w_encWh = takeH((size_t)G3 * H_);
    _Float16* w_mu    = takeH((size_t)Z_ * H_);
    _Float16* w_lv    = takeH((size_t)Z_ * H_);
    _Float16* w_z2h   = takeH((size_t)H_ * Z_);
    _Float16* w_h2o   = takeH((size_t)SYM * H_);
    _Float16* w_ancWi = takeH((size_t)G3 * SYM);
    _Float16* w_ancWh = takeH((size_t)G3 * H_);
    _Float16* w_fraWi = takeH((size_t)G3 * SYM);
    _Float16* w_ua    = takeH((size_t)H_ * H_);
    _Float16* w_uf    = takeH((size_t)H_ * H_);

    _Float16* T16   = takeH((size_t)NN * B_ * SYM);   // targets f16
    _Float16* ENCH  = takeH((size_t)NN * B_ * H_);    // encoder h per node
    _Float16* HAI   = takeH((size_t)15 * B_ * H_);    // decoder ancestral h per internal node
    _Float16* HID   = takeH((size_t)NN * B_ * H_);    // decoder hidden per node
    _Float16* PROBS = takeH((size_t)NN * B_ * SYM);   // softmax probs per node
    _Float16* Z16   = takeH((size_t)B_ * Z_);         // latent
    _Float16* S16   = takeH((size_t)8 * B_ * H_);     // encoder s
    _Float16* HF16  = takeH((size_t)8 * B_ * H_);     // fraternal h_f
    float* SCA = takeF((size_t)16 * B_ * G3);         // 131072 x 384 f32 scratch
    float* SCB = takeF((size_t)8 * B_ * 512);         // 65536 x 512 f32 scratch

    // dispatch: K in {32,128}, NT in {4,2}, ACC/BIAS compile-time
    auto gemm = [&](const _Float16* A, const _Float16* W, const float* bias, float* C,
                    int M, int N, int K, int aStr, int cStr, int acc) {
        dim3 g((unsigned)(N / 64), (unsigned)(M / 128));
        if (K == 32) {
            // all K=32 GEMMs are store+bias, N=384
            k_gemm16<32, 4, false, true><<<g, 256, 0, stream>>>(A, W, bias, C, M, N, aStr, cStr);
        } else if (N <= 32) {
            dim3 g2(1, (unsigned)(M / 128));
            k_gemm16<128, 2, false, true><<<g2, 256, 0, stream>>>(A, W, bias, C, M, N, aStr, cStr);
        } else if (!acc) {
            k_gemm16<128, 4, false, true><<<g, 256, 0, stream>>>(A, W, bias, C, M, N, aStr, cStr);
        } else if (bias) {
            k_gemm16<128, 4, true, true><<<g, 256, 0, stream>>>(A, W, bias, C, M, N, aStr, cStr);
        } else {
            k_gemm16<128, 4, true, false><<<g, 256, 0, stream>>>(A, W, bias, C, M, N, aStr, cStr);
        }
    };
    auto NB = [](long tot) { return dim3((unsigned)((tot + 255) / 256)); };

    // ---- convert weights + targets to f16 ----
    k_f2h<<<NB((long)G3 * SYM), 256, 0, stream>>>(enc_Wi, w_encWi, (long)G3 * SYM);
    k_f2h<<<NB((long)G3 * H_),  256, 0, stream>>>(enc_Wh, w_encWh, (long)G3 * H_);
    k_f2h<<<NB((long)Z_ * H_),  256, 0, stream>>>(mu_W,  w_mu,  (long)Z_ * H_);
    k_f2h<<<NB((long)Z_ * H_),  256, 0, stream>>>(lv_W,  w_lv,  (long)Z_ * H_);
    k_f2h<<<NB((long)H_ * Z_),  256, 0, stream>>>(z2h_W, w_z2h, (long)H_ * Z_);
    k_f2h<<<NB((long)SYM * H_), 256, 0, stream>>>(h2o_W, w_h2o, (long)SYM * H_);
    k_f2h<<<NB((long)G3 * SYM), 256, 0, stream>>>(anc_Wi, w_ancWi, (long)G3 * SYM);
    k_f2h<<<NB((long)G3 * H_),  256, 0, stream>>>(anc_Wh, w_ancWh, (long)G3 * H_);
    k_f2h<<<NB((long)G3 * SYM), 256, 0, stream>>>(fra_Wi, w_fraWi, (long)G3 * SYM);
    k_f2h<<<NB((long)H_ * H_),  256, 0, stream>>>(ua_W, w_ua, (long)H_ * H_);
    k_f2h<<<NB((long)H_ * H_),  256, 0, stream>>>(uf_W, w_uf, (long)H_ * H_);
    k_f2h<<<NB((long)NN * B_ * SYM), 256, 0, stream>>>(targets, T16, (long)NN * B_ * SYM);

    // ================= Encoder (bottom-up, whole level batched) =================
    {   // leaves: nodes 15..30, M = 131072
        int n0 = 15, M = 16 * B_;
        gemm(T16 + (size_t)n0 * B_ * SYM, w_encWi, enc_bi, SCA, M, G3, SYM, B_, B_, 0);
        k_enc_leaf<<<NB((long)M * H_), 256, 0, stream>>>(SCA, enc_bh, masks,
                                                         ENCH + (size_t)n0 * B_ * H_, M, n0);
    }
    for (int L = 3; L >= 0; --L) {
        int n0 = (1 << L) - 1, M = (1 << L) * B_;
        const _Float16* hL = ENCH + (size_t)(2 * n0 + 1) * B_ * H_;  // stride-2 node slabs
        const _Float16* hR = ENCH + (size_t)(2 * n0 + 2) * B_ * H_;
        float* RL = SCB;
        float* RR = SCB + 1 * (size_t)M * H_;
        float* BZ = SCB + 2 * (size_t)M * H_;
        float* BN = SCB + 3 * (size_t)M * H_;
        gemm(T16 + (size_t)n0 * B_ * SYM, w_encWi, enc_bi, SCA, M, G3, SYM, B_, B_, 0);
        gemm(hL, w_encWh,                     enc_bh,      RL, M, H_, H_, 2 * B_, B_, 0); // Whr
        gemm(hR, w_encWh,                     enc_bh,      RR, M, H_, H_, 2 * B_, B_, 0);
        gemm(hL, w_encWh + (size_t)H_ * H_,   enc_bh + H_, BZ, M, H_, H_, 2 * B_, B_, 0); // Whz
        gemm(hR, w_encWh + (size_t)H_ * H_,   nullptr,     BZ, M, H_, H_, 2 * B_, B_, 1); // +=
        k_enc_s<<<NB((long)M * H_), 256, 0, stream>>>(SCA, RL, RR, hL, hR, S16, M, 2 * B_);
        gemm(S16, w_encWh + 2 * (size_t)H_ * H_, enc_bh + 2 * H_, BN, M, H_, H_, B_, B_, 0); // Whn
        k_enc_h<<<NB((long)M * H_), 256, 0, stream>>>(SCA, BZ, BN, hL, hR, masks,
                                                      ENCH + (size_t)n0 * B_ * H_, M, n0, 2 * B_);
    }
    // mu / logvar / reparameterize
    gemm(ENCH, w_mu, mu_b, outMu, B_, Z_, H_, B_, B_, 0);
    gemm(ENCH, w_lv, lv_b, outLv, B_, Z_, H_, B_, B_, 0);
    k_reparam<<<NB((long)B_ * Z_), 256, 0, stream>>>(outMu, outLv, eps, Z16, (long)B_ * Z_);

    // ================= Decoder (top-down, level batched) =================
    gemm(Z16, w_z2h, z2h_b, SCB, B_, H_, Z_, B_, B_, 0);                 // hidden0 f32
    k_f2h<<<NB((long)B_ * H_), 256, 0, stream>>>(SCB, HID, (long)B_ * H_);

    // root: pred, probs, ancestral GRU
    gemm(HID, w_h2o, h2o_b, outPred, B_, SYM, H_, B_, B_, 0);
    k_softmax32<<<NB(B_), 256, 0, stream>>>(outPred, PROBS, B_, B_);
    gemm(PROBS, w_ancWi, anc_bi, SCA, B_, G3, SYM, B_, B_, 0);
    gemm(HID,   w_ancWh, anc_bh, SCB, B_, G3, H_,  B_, B_, 0);
    k_gru<<<NB((long)B_ * H_), 256, 0, stream>>>(SCA, SCB, HID, HAI, B_, B_, B_);

    for (int L = 1; L <= 4; ++L) {
        int p0 = (1 << (L - 1)) - 1;
        int Mp = (1 << (L - 1)) * B_;
        int c1 = 2 * p0 + 1, c2 = 2 * p0 + 2;
        const _Float16* haP = HAI + (size_t)p0 * B_ * H_;                 // dense parents
        _Float16* hidL = HID + (size_t)c1 * B_ * H_;                      // stride-2 slabs
        _Float16* hidR = HID + (size_t)c2 * B_ * H_;
        float*    predL = outPred + (size_t)c1 * B_ * SYM;
        float*    predR = outPred + (size_t)c2 * B_ * SYM;
        _Float16* prL = PROBS + (size_t)c1 * B_ * SYM;
        _Float16* prR = PROBS + (size_t)c2 * B_ * SYM;

        // left child: hidden = parent h_ai ; pred ; probs
        k_copy16<<<NB((long)Mp * H_), 256, 0, stream>>>(haP, hidL, Mp, H_, B_, 2 * B_);
        gemm(hidL, w_h2o, h2o_b, predL, Mp, SYM, H_, 2 * B_, 2 * B_, 0);
        k_softmax32<<<NB(Mp), 256, 0, stream>>>(predL, prL, Mp, 2 * B_);

        // right child hidden: tanh(uf(fraGRU(probs_left, 0)) + ua(h_ai))
        gemm(prL, w_fraWi, fra_bi, SCA, Mp, G3, SYM, 2 * B_, B_, 0);
        k_fra<<<NB((long)Mp * H_), 256, 0, stream>>>(SCA, fra_bh, HF16, Mp);
        gemm(HF16, w_uf, uf_b, SCB, Mp, H_, H_, B_, B_, 0);
        gemm(haP,  w_ua, ua_b, SCB, Mp, H_, H_, B_, B_, 1);               // accumulate
        k_tanh16<<<NB((long)Mp * H_), 256, 0, stream>>>(SCB, hidR, Mp, 2 * B_);
        gemm(hidR, w_h2o, h2o_b, predR, Mp, SYM, H_, 2 * B_, 2 * B_, 0);
        k_softmax32<<<NB(Mp), 256, 0, stream>>>(predR, prR, Mp, 2 * B_);

        if (L < 4) {  // ancestral GRU for this level (left batch, right batch)
            gemm(prL, w_ancWi, anc_bi, SCA, Mp, G3, SYM, 2 * B_, B_, 0);
            gemm(haP, w_ancWh, anc_bh, SCB, Mp, G3, H_,  B_, B_, 0);
            k_gru<<<NB((long)Mp * H_), 256, 0, stream>>>(SCA, SCB, haP,
                HAI + (size_t)c1 * B_ * H_, Mp, B_, 2 * B_);
            gemm(prR, w_ancWi, anc_bi, SCA, Mp, G3, SYM, 2 * B_, B_, 0);
            gemm(haP, w_ancWh, anc_bh, SCB, Mp, G3, H_,  B_, B_, 0);
            k_gru<<<NB((long)Mp * H_), 256, 0, stream>>>(SCA, SCB, haP,
                HAI + (size_t)c2 * B_ * H_, Mp, B_, 2 * B_);
        }
    }
}
